// Actor_att1_66975720013855
// MI455X (gfx1250) — matrix operand order repaired and checked by reading, not compiled, not run on hardware
//
#include <hip/hip_runtime.h>

typedef __attribute__((ext_vector_type(16))) _Float16 v16h;
typedef __attribute__((ext_vector_type(8)))  float    v8f;

#define WMMA(A,B,C) __builtin_amdgcn_wmma_f32_16x16x32_f16(false,(A),false,(B),(short)0,(C),false,false)

#define N_GOOD     16
#define OBS_DIM    127
#define TILE_M     16
#define WAVES_PER_WG 8
#define SCR_STRIDE 50      // bank-conflict-free for both D-store and A-load patterns
#define LN_EPS     1e-5f
#define N_FRAGS    16
#define FRAG_BYTES (N_FRAGS * 32 * (int)sizeof(v16h))   // 16 KB

// ---- cross-lane sum over each 16-lane half (xor 1/2/4/8 never crosses bit 4) ----
__device__ __forceinline__ float rsum16(float v) {
    v += __shfl_xor(v, 1);
    v += __shfl_xor(v, 2);
    v += __shfl_xor(v, 4);
    v += __shfl_xor(v, 8);
    return v;
}

// ---- pack B fragment (KxN tile of row-major w[Krows][Ncols]) for v_wmma_f32_16x16x32_f16 ----
// B layout: lane -> N = lane&15 ; element e -> K = (lane<16 ? 0 : 16) + e (+k_off)
__device__ __forceinline__ v16h packB(const float* __restrict__ w, int Krows, int Ncols,
                                      int k_off, int n_off, int lane) {
    int n  = (lane & 15) + n_off;
    int kb = k_off + ((lane & 16) ? 16 : 0);
    v16h b;
#pragma unroll
    for (int e = 0; e < 16; ++e) {
        int k = kb + e;
        float v = (k < Krows && n < Ncols) ? w[k * Ncols + n] : 0.0f;
        b[e] = (_Float16)v;
    }
    return b;
}

// ---- pack A fragment from per-lane feature array, K <= 8 ----
// A layout: lane -> M = lane&15 ; element e -> K = (lane<16 ? 0 : 16) + e (nonzero only for lane<16, e<K)
template <int K>
__device__ __forceinline__ v16h packA_small(const float* f, int lane) {
    v16h a;
    bool lo = (lane & 16) == 0;
#pragma unroll
    for (int e = 0; e < 8; ++e) {
        float v = (lo && e < K) ? f[e] : 0.0f;
        a[e]     = (_Float16)v;
        a[8 + e] = (_Float16)0.0f;
    }
    return a;
}

// ---- load A fragment from this lane's LDS activation row (row-major, SCR_STRIDE floats) ----
__device__ __forceinline__ v16h loadA_lds(const float* row, int k_off, int Kmax, int lane) {
    int b0 = k_off + ((lane & 16) ? 16 : 0);
    v16h a;
#pragma unroll
    for (int e = 0; e < 8; ++e) {
        int k0 = b0 + e, k1 = b0 + 8 + e;
        a[e]     = (k0 < Kmax) ? (_Float16)row[k0] : (_Float16)0.0f;
        a[8 + e] = (k1 < Kmax) ? (_Float16)row[k1] : (_Float16)0.0f;
    }
    return a;
}

// ---- store a 16x16 D tile to LDS at column offset (C/D layout: M = r + 8*(lane>=16), N = lane&15) ----
__device__ __forceinline__ void storeD16(float* scr, const v8f& d, int col_off, int lane) {
    int n  = (lane & 15) + col_off;
    int mb = (lane & 16) ? 8 : 0;
#pragma unroll
    for (int r = 0; r < 8; ++r) scr[(mb + r) * SCR_STRIDE + n] = d[r];
}

// ---- LayerNorm(16) * g + b, relu — operates in C/D layout ----
__device__ __forceinline__ v8f ln_relu(const v8f& x, const float* g, const float* bln, int lane) {
    int nD = lane & 15;
    float gg = g[nD], bb = bln[nD];
    v8f y;
#pragma unroll
    for (int r = 0; r < 8; ++r) {
        float mu  = rsum16(x[r]) * 0.0625f;
        float d   = x[r] - mu;
        float var = rsum16(d * d) * 0.0625f;
        float v   = d * rsqrtf(var + LN_EPS) * gg + bb;
        y[r] = fmaxf(v, 0.0f);
    }
    return y;
}

// ---- entity encoder (FEAT->32->16, relu) + online-softmax attention update ----
template <int FEAT>
__device__ __forceinline__ void encode_attend(const float* f,
                                              const v16h& w1t0, const v16h& w1t1, const float* b1,
                                              const v16h& w2t, const float* b2,
                                              const v8f& self_d, v8f& acc, float* sm,
                                              float* scr, const float* scr_rowA, int lane) {
    v8f zero = {};
    int nD = lane & 15;
    v16h a  = packA_small<FEAT>(f, lane);
    v8f h0  = WMMA(a, w1t0, zero);
    v8f h1  = WMMA(a, w1t1, zero);
    float bb0 = b1[nD], bb1 = b1[nD + 16];
#pragma unroll
    for (int r = 0; r < 8; ++r) {
        h0[r] = fmaxf(h0[r] + bb0, 0.0f);
        h1[r] = fmaxf(h1[r] + bb1, 0.0f);
    }
    storeD16(scr, h0, 0, lane);
    storeD16(scr, h1, 16, lane);
    __builtin_amdgcn_wave_barrier();          // keep DS store->load order (in-order per wave)
    v16h ah = loadA_lds(scr_rowA, 0, 32, lane);
    __builtin_amdgcn_wave_barrier();
    v8f enc = WMMA(ah, w2t, zero);
    float bb2 = b2[nD];
#pragma unroll
    for (int r = 0; r < 8; ++r) enc[r] = fmaxf(enc[r] + bb2, 0.0f);
#pragma unroll
    for (int r = 0; r < 8; ++r) {
        float s = rsum16(self_d[r] * enc[r]) * 0.25f;
        float e = __expf(s);
        sm[r]  += e;
        acc[r] += e * enc[r];
    }
}

// ---- one-wave setup kernel: pack all weight B-fragments once into workspace ----
// frag layout in d_ws: frag f, lane l -> v16h at wf[f*32 + l]
__global__ void pack_weights_kernel(const float* en_w1, const float* en_w2,
                                    const float* oa_w1, const float* oa_w2,
                                    const float* g_w1,  const float* g_w2,
                                    const float* m_w1,  const float* m_w2, const float* m_w3,
                                    v16h* __restrict__ wf) {
    if (threadIdx.x >= 32) return;
    int lane = threadIdx.x;
    wf[ 0 * 32 + lane] = packB(en_w1, 4, 32, 0, 0, lane);
    wf[ 1 * 32 + lane] = packB(en_w1, 4, 32, 0, 16, lane);
    wf[ 2 * 32 + lane] = packB(en_w2, 32, 16, 0, 0, lane);
    wf[ 3 * 32 + lane] = packB(oa_w1, 5, 32, 0, 0, lane);
    wf[ 4 * 32 + lane] = packB(oa_w1, 5, 32, 0, 16, lane);
    wf[ 5 * 32 + lane] = packB(oa_w2, 32, 16, 0, 0, lane);
    wf[ 6 * 32 + lane] = packB(g_w1, 3, 32, 0, 0, lane);
    wf[ 7 * 32 + lane] = packB(g_w1, 3, 32, 0, 16, lane);
    wf[ 8 * 32 + lane] = packB(g_w2, 32, 16, 0, 0, lane);
    wf[ 9 * 32 + lane] = packB(m_w1, 48, 32, 0, 0, lane);
    wf[10 * 32 + lane] = packB(m_w1, 48, 32, 0, 16, lane);
    wf[11 * 32 + lane] = packB(m_w1, 48, 32, 32, 0, lane);
    wf[12 * 32 + lane] = packB(m_w1, 48, 32, 32, 16, lane);
    wf[13 * 32 + lane] = packB(m_w2, 32, 32, 0, 0, lane);
    wf[14 * 32 + lane] = packB(m_w2, 32, 32, 0, 16, lane);
    wf[15 * 32 + lane] = packB(m_w3, 32, 2, 0, 0, lane);
}

// PRE=true: load prepacked fragments (2x global_load_b128, L2 broadcast)
// PRE=false: self-contained fallback (inline packing) when ws is too small
template <bool PRE>
__device__ __forceinline__ v16h getB(const v16h* __restrict__ wf, int frag,
                                     const float* __restrict__ w, int Krows, int Ncols,
                                     int k_off, int n_off, int lane) {
    if (PRE) return wf[frag * 32 + lane];
    return packB(w, Krows, Ncols, k_off, n_off, lane);
}

template <bool PRE>
__global__ __launch_bounds__(WAVES_PER_WG * 32)
void actor_att_kernel(const float* __restrict__ s_input,
                      const float* en_w1, const float* en_b1, const float* en_w2, const float* en_b2,
                      const float* oa_w1, const float* oa_b1, const float* oa_w2, const float* oa_b2,
                      const float* oa_g, const float* oa_bln,
                      const float* g_w1, const float* g_b1, const float* g_w2, const float* g_b2,
                      const float* g_g, const float* g_bln,
                      const float* m_w1, const float* m_b1, const float* m_w2, const float* m_b2,
                      const float* m_w3, const float* m_b3,
                      const v16h* __restrict__ wf,
                      float* __restrict__ out, int n_rows) {
    __shared__ float scr_all[WAVES_PER_WG][TILE_M * SCR_STRIDE];
    int lane = threadIdx.x & 31;
    int wave = threadIdx.x >> 5;
    float* scr = scr_all[wave];

    int tile = blockIdx.x * WAVES_PER_WG + wave;
    long tile_row0 = (long)tile * TILE_M;
    if (tile_row0 >= n_rows) return;          // wave-uniform: EXEC stays all-1s for WMMA

    int  m_a  = lane & 15;                    // this lane's A-matrix row
    long rowA = tile_row0 + m_a; if (rowA >= n_rows) rowA = n_rows - 1;
    const float* rp = s_input + rowA * OBS_DIM;
    const float* scr_rowA = scr + m_a * SCR_STRIDE;

    int nD  = lane & 15;                      // C/D column
    int mbD = (lane & 16) ? 8 : 0;            // C/D row base
    v8f zero = {};

    // ---------------- self encoder: 4 -> 32 -> 16 ----------------
    v8f self_d;
    {
        float f4[4] = { rp[0], rp[1], rp[2], rp[3] };
        v16h a  = packA_small<4>(f4, lane);
        v16h b0 = getB<PRE>(wf, 0, en_w1, 4, 32, 0, 0, lane);
        v16h b1 = getB<PRE>(wf, 1, en_w1, 4, 32, 0, 16, lane);
        v8f h0 = WMMA(a, b0, zero);
        v8f h1 = WMMA(a, b1, zero);
        float bb0 = en_b1[nD], bb1 = en_b1[nD + 16];
#pragma unroll
        for (int r = 0; r < 8; ++r) {
            h0[r] = fmaxf(h0[r] + bb0, 0.0f);
            h1[r] = fmaxf(h1[r] + bb1, 0.0f);
        }
        storeD16(scr, h0, 0, lane);
        storeD16(scr, h1, 16, lane);
        __builtin_amdgcn_wave_barrier();
        v16h ah = loadA_lds(scr_rowA, 0, 32, lane);
        __builtin_amdgcn_wave_barrier();
        v16h b2 = getB<PRE>(wf, 2, en_w2, 32, 16, 0, 0, lane);
        self_d = WMMA(ah, b2, zero);
        float bb2 = en_b2[nD];
#pragma unroll
        for (int r = 0; r < 8; ++r) self_d[r] = fmaxf(self_d[r] + bb2, 0.0f);
    }

    // ---------------- other agents: 15 x (5 -> 32 -> 16) + attention ----------------
    v8f other_out;
    {
        v16h w1t0 = getB<PRE>(wf, 3, oa_w1, 5, 32, 0, 0, lane);
        v16h w1t1 = getB<PRE>(wf, 4, oa_w1, 5, 32, 0, 16, lane);
        v16h w2t  = getB<PRE>(wf, 5, oa_w2, 32, 16, 0, 0, lane);
        v8f acc = {};
        float sm[8] = {0.f,0.f,0.f,0.f,0.f,0.f,0.f,0.f};
        for (int j = 0; j < N_GOOD - 1; ++j) {
            float f[5] = { rp[4 + 2*j], rp[5 + 2*j], rp[34 + 2*j], rp[35 + 2*j], rp[64 + j] };
            encode_attend<5>(f, w1t0, w1t1, oa_b1, w2t, oa_b2, self_d, acc, sm, scr, scr_rowA, lane);
        }
        v8f att;
#pragma unroll
        for (int r = 0; r < 8; ++r) att[r] = acc[r] / sm[r];
        other_out = ln_relu(att, oa_g, oa_bln, lane);
    }

    // ---------------- food: 16 x (3 -> 32 -> 16) + attention ----------------
    v8f food_out;
    {
        v16h w1t0 = getB<PRE>(wf, 6, g_w1, 3, 32, 0, 0, lane);
        v16h w1t1 = getB<PRE>(wf, 7, g_w1, 3, 32, 0, 16, lane);
        v16h w2t  = getB<PRE>(wf, 8, g_w2, 32, 16, 0, 0, lane);
        v8f acc = {};
        float sm[8] = {0.f,0.f,0.f,0.f,0.f,0.f,0.f,0.f};
        for (int j = 0; j < N_GOOD; ++j) {
            float f[3] = { rp[79 + 3*j], rp[80 + 3*j], rp[81 + 3*j] };
            encode_attend<3>(f, w1t0, w1t1, g_b1, w2t, g_b2, self_d, acc, sm, scr, scr_rowA, lane);
        }
        v8f att;
#pragma unroll
        for (int r = 0; r < 8; ++r) att[r] = acc[r] / sm[r];
        food_out = ln_relu(att, g_g, g_bln, lane);
    }

    // ---------------- merged head: 48 -> 32 -> 32 -> 2 ----------------
    storeD16(scr, self_d,   0, lane);
    storeD16(scr, food_out, 16, lane);
    storeD16(scr, other_out, 32, lane);
    __builtin_amdgcn_wave_barrier();
    v16h a0 = loadA_lds(scr_rowA, 0, 48, lane);
    v16h a1 = loadA_lds(scr_rowA, 32, 48, lane);
    __builtin_amdgcn_wave_barrier();

    v8f h0, h1;
    {
        v16h b00 = getB<PRE>(wf,  9, m_w1, 48, 32, 0, 0, lane);
        v16h b01 = getB<PRE>(wf, 10, m_w1, 48, 32, 0, 16, lane);
        v16h b10 = getB<PRE>(wf, 11, m_w1, 48, 32, 32, 0, lane);
        v16h b11 = getB<PRE>(wf, 12, m_w1, 48, 32, 32, 16, lane);
        h0 = WMMA(a0, b00, zero); h0 = WMMA(a1, b10, h0);
        h1 = WMMA(a0, b01, zero); h1 = WMMA(a1, b11, h1);
        float bb0 = m_b1[nD], bb1 = m_b1[nD + 16];
#pragma unroll
        for (int r = 0; r < 8; ++r) {
            float x0 = h0[r] + bb0, x1 = h1[r] + bb1;
            h0[r] = (x0 > 0.0f) ? x0 : 0.01f * x0;
            h1[r] = (x1 > 0.0f) ? x1 : 0.01f * x1;
        }
    }
    storeD16(scr, h0, 0, lane);
    storeD16(scr, h1, 16, lane);
    __builtin_amdgcn_wave_barrier();
    v16h ah = loadA_lds(scr_rowA, 0, 32, lane);
    __builtin_amdgcn_wave_barrier();
    {
        v16h b0 = getB<PRE>(wf, 13, m_w2, 32, 32, 0, 0, lane);
        v16h b1 = getB<PRE>(wf, 14, m_w2, 32, 32, 0, 16, lane);
        h0 = WMMA(ah, b0, zero);
        h1 = WMMA(ah, b1, zero);
        float bb0 = m_b2[nD], bb1 = m_b2[nD + 16];
#pragma unroll
        for (int r = 0; r < 8; ++r) {
            float x0 = h0[r] + bb0, x1 = h1[r] + bb1;
            h0[r] = (x0 > 0.0f) ? x0 : 0.01f * x0;
            h1[r] = (x1 > 0.0f) ? x1 : 0.01f * x1;
        }
    }
    storeD16(scr, h0, 0, lane);
    storeD16(scr, h1, 16, lane);
    __builtin_amdgcn_wave_barrier();
    ah = loadA_lds(scr_rowA, 0, 32, lane);
    {
        v16h b3 = getB<PRE>(wf, 15, m_w3, 32, 2, 0, 0, lane);
        v8f o = WMMA(ah, b3, zero);
        if (nD < 2) {
            float bb = m_b3[nD];
#pragma unroll
            for (int r = 0; r < 8; ++r) {
                long row = tile_row0 + mbD + r;
                if (row < n_rows) out[row * 2 + nD] = tanhf(o[r] + bb);
            }
        }
    }
}

extern "C" void kernel_launch(void* const* d_in, const int* in_sizes, int n_in,
                              void* d_out, int out_size, void* d_ws, size_t ws_size,
                              hipStream_t stream) {
    (void)n_in; (void)out_size;
    const float* s_input = (const float*)d_in[0];
    const float* en_w1 = (const float*)d_in[1];
    const float* en_b1 = (const float*)d_in[2];
    const float* en_w2 = (const float*)d_in[3];
    const float* en_b2 = (const float*)d_in[4];
    const float* oa_w1 = (const float*)d_in[5];
    const float* oa_b1 = (const float*)d_in[6];
    const float* oa_w2 = (const float*)d_in[7];
    const float* oa_b2 = (const float*)d_in[8];
    const float* oa_g  = (const float*)d_in[9];
    const float* oa_bln= (const float*)d_in[10];
    const float* g_w1  = (const float*)d_in[11];
    const float* g_b1  = (const float*)d_in[12];
    const float* g_w2  = (const float*)d_in[13];
    const float* g_b2  = (const float*)d_in[14];
    const float* g_g   = (const float*)d_in[15];
    const float* g_bln = (const float*)d_in[16];
    const float* m_w1  = (const float*)d_in[17];
    const float* m_b1  = (const float*)d_in[18];
    const float* m_w2  = (const float*)d_in[19];
    const float* m_b2  = (const float*)d_in[20];
    const float* m_w3  = (const float*)d_in[21];
    const float* m_b3  = (const float*)d_in[22];

    int n_rows = in_sizes[0] / OBS_DIM;
    int tiles  = (n_rows + TILE_M - 1) / TILE_M;
    int blocks = (tiles + WAVES_PER_WG - 1) / WAVES_PER_WG;

    if (ws_size >= (size_t)FRAG_BYTES) {
        v16h* wf = (v16h*)d_ws;
        pack_weights_kernel<<<1, 32, 0, stream>>>(en_w1, en_w2, oa_w1, oa_w2,
                                                  g_w1, g_w2, m_w1, m_w2, m_w3, wf);
        actor_att_kernel<true><<<blocks, WAVES_PER_WG * 32, 0, stream>>>(
            s_input, en_w1, en_b1, en_w2, en_b2,
            oa_w1, oa_b1, oa_w2, oa_b2, oa_g, oa_bln,
            g_w1, g_b1, g_w2, g_b2, g_g, g_bln,
            m_w1, m_b1, m_w2, m_b2, m_w3, m_b3,
            wf, (float*)d_out, n_rows);
    } else {
        actor_att_kernel<false><<<blocks, WAVES_PER_WG * 32, 0, stream>>>(
            s_input, en_w1, en_b1, en_w2, en_b2,
            oa_w1, oa_b1, oa_w2, oa_b2, oa_g, oa_bln,
            g_w1, g_b1, g_w2, g_b2, g_g, g_bln,
            m_w1, m_b1, m_w2, m_b2, m_w3, m_b3,
            (const v16h*)nullptr, (float*)d_out, n_rows);
    }
}
